// RIMCell_31671088840704
// MI455X (gfx1250) — matrix operand order repaired and checked by reading, not compiled, not run on hardware
//
#include <hip/hip_runtime.h>
#include <hip/hip_bf16.h>

// MI455X (gfx1250) RIM-cell forward.
// All large GEMMs use V_WMMA_F32_16X16X32_BF16 (bf16 A/B, fp32 accumulate):
// the problem is HBM-bound (~135MB @ 23.3TB/s ~ 5.8us) only if matrix math
// runs at the bf16 WMMA rate; f32 16x16x4 WMMA (K=4/issue) would be ~9x
// compute-bound.  Global->LDS staging uses a register-prefetch double buffer
// so global_load_b128 latency is hidden behind the WMMA stream (an async/TDM
// copy cannot be used: the fp32->bf16 conversion must happen in flight).

typedef __attribute__((ext_vector_type(16))) __bf16 v16bf;
typedef __attribute__((ext_vector_type(8)))  float  v8f;

union WFrag {
    uint4 q[2];
    v16bf v;
};

#define BM 64
#define BN 128
#define BK 32
#define APAD 40   // padded LDS row stride (bf16 elems), keeps 16B alignment

// ---------------------------------------------------------------------------
// Generic batched GEMM:  C[M,N] = A[M,K] @ B[K,N] (+ bias[N]), fp32 in/out,
// bf16 WMMA accumulate-in-fp32.  Row-major A (lda), row-major B (ldb),
// row-major C (ldc).  blockIdx.z selects batch slice via element strides.
// Requirements (true for every call below): K % 32 == 0, N % 16 == 0,
// M % 64 == 0 (no M guard needed).
// ---------------------------------------------------------------------------
__global__ __launch_bounds__(256)
void rim_gemm_bf16_wmma(const float* __restrict__ A, int lda, long long sA,
                        const float* __restrict__ Bw, int ldb, long long sB,
                        const float* __restrict__ bias, long long sBias,
                        float* __restrict__ C, int ldc, long long sC,
                        int M, int N, int Kdim)
{
    __shared__ __bf16 Asm[BM][APAD];
    __shared__ __bf16 Bsm[BN][APAD];

    const int u = blockIdx.z;
    A  += (long long)u * sA;
    Bw += (long long)u * sB;
    C  += (long long)u * sC;
    if (bias) bias += (long long)u * sBias;

    const int m0 = blockIdx.y * BM;
    const int n0 = blockIdx.x * BN;

    const int tid  = threadIdx.x;
    const int w    = tid >> 5;          // wave id 0..7
    const int lane = tid & 31;
    const int half = lane >> 4;         // 0/1
    const int l15  = lane & 15;
    const int wm   = w >> 2;            // 0..1  (row group of 32)
    const int wn   = w & 3;             // 0..3  (col group of 32)

    // Cooperative-load geometry (k0-invariant).
    const int aRow = tid >> 3;          // + r*32, r=0..1
    const int aCol = (tid & 7) * 4;     // float4 column within 32-wide tile
    const int bRow = tid >> 5;          // + r*8,  r=0..3 (k-row)
    const int gN   = n0 + (tid & 31) * 4;
    const int gNc  = (gN < N) ? gN : 0;     // clamped address, no branch
    const bool nOK = (gN < N);              // N%4==0 -> whole float4 valid

    v8f acc[2][2] = {};
    float4 va[2], vb[4];

    // ---- prologue: fetch tile 0 ----
    #pragma unroll
    for (int r = 0; r < 2; r++)
        va[r] = *(const float4*)(A + (long long)(m0 + aRow + r * 32) * lda + aCol);
    #pragma unroll
    for (int r = 0; r < 4; r++) {
        float4 v4 = *(const float4*)(Bw + (long long)(bRow + r * 8) * ldb + gNc);
        if (!nOK) v4 = make_float4(0.f, 0.f, 0.f, 0.f);
        vb[r] = v4;
    }
    #pragma unroll
    for (int r = 0; r < 2; r++) {
        __bf16* dst = &Asm[aRow + r * 32][aCol];
        dst[0] = (__bf16)va[r].x; dst[1] = (__bf16)va[r].y;
        dst[2] = (__bf16)va[r].z; dst[3] = (__bf16)va[r].w;
    }
    #pragma unroll
    for (int r = 0; r < 4; r++) {
        int krow = bRow + r * 8, c = (tid & 31) * 4;
        Bsm[c + 0][krow] = (__bf16)vb[r].x;
        Bsm[c + 1][krow] = (__bf16)vb[r].y;
        Bsm[c + 2][krow] = (__bf16)vb[r].z;
        Bsm[c + 3][krow] = (__bf16)vb[r].w;
    }

    for (int k0 = 0; k0 < Kdim; k0 += BK) {
        __syncthreads();                       // LDS tile ready

        // ---- prefetch next tile into registers (overlaps the WMMAs) ----
        const bool more = (k0 + BK) < Kdim;
        if (more) {
            const int kn = k0 + BK;
            #pragma unroll
            for (int r = 0; r < 2; r++)
                va[r] = *(const float4*)(A + (long long)(m0 + aRow + r * 32) * lda + kn + aCol);
            #pragma unroll
            for (int r = 0; r < 4; r++) {
                float4 v4 = *(const float4*)(Bw + (long long)(kn + bRow + r * 8) * ldb + gNc);
                if (!nOK) v4 = make_float4(0.f, 0.f, 0.f, 0.f);
                vb[r] = v4;
            }
        }

        // ---- fragments per documented 16-bit A/B layouts ----
        WFrag a[2], b[2];
        #pragma unroll
        for (int mi = 0; mi < 2; mi++) {
            // lane<16: M=l15, K = {0..7, 16..23}; lane>=16: K = {8..15, 24..31}
            const __bf16* ap = &Asm[wm * 32 + mi * 16 + l15][8 * half];
            a[mi].q[0] = *(const uint4*)ap;          // K = 8*half   .. +7
            a[mi].q[1] = *(const uint4*)(ap + 16);   // K = 16+8*half.. +7
        }
        #pragma unroll
        for (int ni = 0; ni < 2; ni++) {
            // lane<16: N=l15, K = {0..7, 16..23}; lane>=16: K = {8..15, 24..31}
            const __bf16* bp = &Bsm[wn * 32 + ni * 16 + l15][8 * half];
            b[ni].q[0] = *(const uint4*)bp;
            b[ni].q[1] = *(const uint4*)(bp + 16);
        }
        #pragma unroll
        for (int mi = 0; mi < 2; mi++)
            #pragma unroll
            for (int ni = 0; ni < 2; ni++)
                acc[mi][ni] = __builtin_amdgcn_wmma_f32_16x16x32_bf16(
                    false, a[mi].v, false, b[ni].v,
                    (short)0, acc[mi][ni], false, false);

        __syncthreads();                       // everyone done reading LDS

        // ---- commit the prefetched tile to LDS ----
        if (more) {
            #pragma unroll
            for (int r = 0; r < 2; r++) {
                __bf16* dst = &Asm[aRow + r * 32][aCol];
                dst[0] = (__bf16)va[r].x; dst[1] = (__bf16)va[r].y;
                dst[2] = (__bf16)va[r].z; dst[3] = (__bf16)va[r].w;
            }
            #pragma unroll
            for (int r = 0; r < 4; r++) {
                int krow = bRow + r * 8, c = (tid & 31) * 4;
                Bsm[c + 0][krow] = (__bf16)vb[r].x;
                Bsm[c + 1][krow] = (__bf16)vb[r].y;
                Bsm[c + 2][krow] = (__bf16)vb[r].z;
                Bsm[c + 3][krow] = (__bf16)vb[r].w;
            }
        }
    }

    // ---- bias + writeback (C/D layout: VGPR j -> M=j (+8 for hi half)) ----
    #pragma unroll
    for (int mi = 0; mi < 2; mi++) {
        #pragma unroll
        for (int ni = 0; ni < 2; ni++) {
            int ccol = n0 + wn * 32 + ni * 16 + l15;
            if (ccol < N) {
                float bv = bias ? bias[ccol] : 0.0f;
                #pragma unroll
                for (int j = 0; j < 8; j++) {
                    int crow = m0 + wm * 32 + mi * 16 + half * 8 + j;
                    C[(long long)crow * ldc + ccol] = acc[mi][ni][j] + bv;
                }
            }
        }
    }
}

// ---------------------------------------------------------------------------
// Input attention: scores over {key0, key1=bk_in}, top-4-of-8 mask,
// 2-way softmax, inputs = mask * (p0*val0 + p1*bv_in).  One block per batch.
// ---------------------------------------------------------------------------
__global__ __launch_bounds__(256)
void rim_scores(const float* __restrict__ qry,   // (B,8,64)
                const float* __restrict__ key0,  // (B,64)
                const float* __restrict__ bk_in, // (64)
                const float* __restrict__ val0,  // (B,512)
                const float* __restrict__ bv_in, // (512)
                float* __restrict__ mask,        // (B,8)
                float* __restrict__ inputs)      // (B,8,512)
{
    const int b   = blockIdx.x;
    const int tid = threadIdx.x;
    const int w   = tid >> 5;
    const int ln  = tid & 31;
    __shared__ float s0[8], s1[8], sm[8], p0s[8], p1s[8];

    // wave w computes the two 64-length dots for unit u = w
    const float* qp = qry + ((long long)b * 8 + w) * 64;
    const float* kp = key0 + (long long)b * 64;
    float d0 = 0.f, d1 = 0.f;
    for (int i = ln; i < 64; i += 32) {
        float qv = qp[i];
        d0 += qv * kp[i];
        d1 += qv * bk_in[i];
    }
    #pragma unroll
    for (int off = 16; off > 0; off >>= 1) {
        d0 += __shfl_down(d0, off, 32);
        d1 += __shfl_down(d1, off, 32);
    }
    if (ln == 0) { s0[w] = d0 * 0.125f; s1[w] = d1 * 0.125f; }  // /sqrt(64)
    __syncthreads();

    if (tid < 8) {
        int u = tid;
        float su = s0[u];
        int rank = 0;
        for (int j = 0; j < 8; j++) {
            float sj = s0[j];
            if (sj > su || (sj == su && j < u)) rank++;   // jax top_k tie rule
        }
        float m = (rank < 4) ? 1.f : 0.f;
        sm[u] = m;
        mask[(long long)b * 8 + u] = m;
        float a = s0[u], c = s1[u];
        float mx = fmaxf(a, c);
        float e0 = __expf(a - mx), e1 = __expf(c - mx);
        float inv = 1.f / (e0 + e1);
        p0s[u] = e0 * inv;
        p1s[u] = e1 * inv;
    }
    __syncthreads();

    const float* vp = val0 + (long long)b * 512;
    for (int u = 0; u < 8; u++) {
        float m = sm[u], p0 = p0s[u], p1 = p1s[u];
        for (int i = tid; i < 512; i += 256)
            inputs[((long long)b * 8 + u) * 512 + i] = m * (p0 * vp[i] + p1 * bv_in[i]);
    }
}

// ---------------------------------------------------------------------------
// GRU gates: hs_new = (1-z)*n + z*hs
// ---------------------------------------------------------------------------
__global__ __launch_bounds__(256)
void rim_gates(const float* __restrict__ gi, const float* __restrict__ gh,
               const float* __restrict__ hs, float* __restrict__ hsnew)
{
    long long idx = (long long)blockIdx.x * 256 + threadIdx.x;   // B*8*512
    if (idx >= 512LL * 8 * 512) return;
    long long bu = idx >> 9;
    int h = (int)(idx & 511);
    long long g = bu * 1536;
    float ir = gi[g + h],        hr = gh[g + h];
    float iz = gi[g + 512 + h],  hz = gh[g + 512 + h];
    float in_ = gi[g + 1024 + h], hn = gh[g + 1024 + h];
    float r = 1.f / (1.f + __expf(-(ir + hr)));
    float z = 1.f / (1.f + __expf(-(iz + hz)));
    float n = tanhf(in_ + r * hn);
    hsnew[idx] = (1.f - z) * n + z * hs[idx];
}

// ---------------------------------------------------------------------------
// Communication attention: per (b, head) 8x8 attention over units,
// ctx[b,u,ch*512+d] = sum_n softmax(q.k/sqrt(32))[u,n]*mask[u] * v[b,n,ch*512+d]
// ---------------------------------------------------------------------------
__global__ __launch_bounds__(256)
void rim_attn(const float* __restrict__ qbuf,  // (B,8,128)
              const float* __restrict__ kbuf,  // (B,8,128)
              const float* __restrict__ vbuf,  // (B,8,2048)
              const float* __restrict__ mask,  // (B,8)
              float* __restrict__ ctx)         // (B,8,2048)
{
    const int b  = blockIdx.x >> 2;
    const int ch = blockIdx.x & 3;
    const int tid = threadIdx.x;
    __shared__ float qs[8][32], ks[8][32], vs[8][512], ps[8][8];

    {   // q/k tiles: 8*32 = 256 elements each
        int u = tid >> 5, kk = tid & 31;
        qs[u][kk] = qbuf[((long long)b * 8 + u) * 128 + ch * 32 + kk];
        ks[u][kk] = kbuf[((long long)b * 8 + u) * 128 + ch * 32 + kk];
    }
    for (int i = tid; i < 4096; i += 256) {      // v tile 8x512
        int n = i >> 9, d = i & 511;
        vs[n][d] = vbuf[((long long)b * 8 + n) * 2048 + ch * 512 + d];
    }
    __syncthreads();

    if (tid < 64) {
        int u = tid >> 3, n = tid & 7;
        float s = 0.f;
        #pragma unroll
        for (int kk = 0; kk < 32; kk++) s += qs[u][kk] * ks[n][kk];
        ps[u][n] = s * 0.17677669529663687f;     // 1/sqrt(32)
    }
    __syncthreads();

    if (tid < 8) {
        int u = tid;
        float mx = -1e30f;
        for (int n = 0; n < 8; n++) mx = fmaxf(mx, ps[u][n]);
        float e[8], sum = 0.f;
        for (int n = 0; n < 8; n++) { e[n] = __expf(ps[u][n] - mx); sum += e[n]; }
        float inv = mask[(long long)b * 8 + u] / sum;
        for (int n = 0; n < 8; n++) ps[u][n] = e[n] * inv;
    }
    __syncthreads();

    for (int i = tid; i < 4096; i += 256) {
        int u = i >> 9, d = i & 511;
        float s = 0.f;
        #pragma unroll
        for (int n = 0; n < 8; n++) s += ps[u][n] * vs[n][d];
        ctx[((long long)b * 8 + u) * 2048 + ch * 512 + d] = s;
    }
}

// ---------------------------------------------------------------------------
// Final merge: out = mask ? hs_new + ctx_proj : hs
// ---------------------------------------------------------------------------
__global__ __launch_bounds__(256)
void rim_final(const float* __restrict__ hsnew, const float* __restrict__ co,
               const float* __restrict__ hs, const float* __restrict__ mask,
               float* __restrict__ out)
{
    long long idx = (long long)blockIdx.x * 256 + threadIdx.x;
    if (idx >= 512LL * 8 * 512) return;
    long long bu = idx >> 9;
    out[idx] = (mask[bu] > 0.5f) ? (hsnew[idx] + co[idx]) : hs[idx];
}

// ---------------------------------------------------------------------------
static inline void launch_gemm(hipStream_t s,
                               const float* A, int lda, long long sA,
                               const float* B, int ldb, long long sB,
                               const float* bias, long long sBias,
                               float* C, int ldc, long long sC,
                               int M, int N, int K, int batch)
{
    dim3 grid((N + BN - 1) / BN, (M + BM - 1) / BM, batch);
    hipLaunchKernelGGL(rim_gemm_bf16_wmma, grid, dim3(256), 0, s,
                       A, lda, sA, B, ldb, sB, bias, sBias, C, ldc, sC, M, N, K);
}

extern "C" void kernel_launch(void* const* d_in, const int* in_sizes, int n_in,
                              void* d_out, int out_size, void* d_ws, size_t ws_size,
                              hipStream_t stream)
{
    const float* x     = (const float*)d_in[0];   // (512,1,1024)
    const float* hs    = (const float*)d_in[1];   // (512,8,512)
    const float* Wk_in = (const float*)d_in[2];   // (1024,64)
    const float* bk_in = (const float*)d_in[3];   // (64)
    const float* Wv_in = (const float*)d_in[4];   // (1024,512)
    const float* bv_in = (const float*)d_in[5];   // (512)
    const float* Wq_in = (const float*)d_in[6];   // (512,64)
    const float* bq_in = (const float*)d_in[7];   // (64)
    const float* W_ih  = (const float*)d_in[8];   // (8,512,1536)
    const float* W_hh  = (const float*)d_in[9];   // (8,512,1536)
    const float* b_ih  = (const float*)d_in[10];  // (8,1536)
    const float* b_hh  = (const float*)d_in[11];  // (8,1536)
    const float* Wq_c  = (const float*)d_in[12];  // (8,512,128)
    const float* Wk_c  = (const float*)d_in[13];  // (8,512,128)
    const float* Wv_c  = (const float*)d_in[14];  // (8,512,2048)
    const float* Wo_c  = (const float*)d_in[15];  // (8,2048,512)
    float* out = (float*)d_out;                   // (512,8,512)
    float* ws  = (float*)d_ws;

    // Workspace layout with lifetime reuse (25,169,920 floats ~ 96 MB):
    float* mask   = ws + 0;          //     4,096
    float* hsnew  = ws + 4096;       // 2,097,152 (live to the end)
    float* gi     = ws + 2101248;    // 6,291,456 } region1
    float* gh     = ws + 8392704;    // 6,291,456 }
    float* vbuf   = ws + 2101248;    // 8,388,608  (reuses gi/gh after gates)
    float* qbuf   = ws + 10489856;   //   524,288
    float* kbuf   = ws + 11014144;   //   524,288
    float* key0   = ws + 14684160;   //    32,768 } region2
    float* val0   = ws + 14716928;   //   262,144 }
    float* qry    = ws + 14979072;   //   262,144 }
    float* inputs = ws + 15241216;   // 2,097,152 }
    float* ctxb   = ws + 14684160;   // 8,388,608  (reuses region2 after gi)
    float* co     = ws + 23072768;   // 2,097,152
    (void)in_sizes; (void)n_in; (void)out_size; (void)ws_size;

    // ---- input attention projections ----
    launch_gemm(stream, x, 1024, 0, Wk_in, 64, 0, bk_in, 0,
                key0, 64, 0, 512, 64, 1024, 1);
    launch_gemm(stream, x, 1024, 0, Wv_in, 512, 0, bv_in, 0,
                val0, 512, 0, 512, 512, 1024, 1);
    launch_gemm(stream, hs, 512, 0, Wq_in, 64, 0, bq_in, 0,
                qry, 64, 0, 4096, 64, 512, 1);
    hipLaunchKernelGGL(rim_scores, dim3(512), dim3(256), 0, stream,
                       qry, key0, bk_in, val0, bv_in, mask, inputs);

    // ---- GRU gate GEMMs (batched over 8 units) ----
    launch_gemm(stream, inputs, 4096, 512, W_ih, 1536, 786432, b_ih, 1536,
                gi, 12288, 1536, 512, 1536, 512, 8);
    launch_gemm(stream, hs, 4096, 512, W_hh, 1536, 786432, b_hh, 1536,
                gh, 12288, 1536, 512, 1536, 512, 8);
    hipLaunchKernelGGL(rim_gates, dim3(8192), dim3(256), 0, stream,
                       gi, gh, hs, hsnew);

    // ---- communication attention projections ----
    launch_gemm(stream, hsnew, 4096, 512, Wq_c, 128, 65536, nullptr, 0,
                qbuf, 1024, 128, 512, 128, 512, 8);
    launch_gemm(stream, hsnew, 4096, 512, Wk_c, 128, 65536, nullptr, 0,
                kbuf, 1024, 128, 512, 128, 512, 8);
    launch_gemm(stream, hsnew, 4096, 512, Wv_c, 2048, 1048576, nullptr, 0,
                vbuf, 16384, 2048, 512, 2048, 512, 8);
    hipLaunchKernelGGL(rim_attn, dim3(2048), dim3(256), 0, stream,
                       qbuf, kbuf, vbuf, mask, ctxb);

    // ---- output projection + masked residual merge ----
    launch_gemm(stream, ctxb, 16384, 2048, Wo_c, 512, 1048576, nullptr, 0,
                co, 4096, 512, 512, 512, 2048, 8);
    hipLaunchKernelGGL(rim_final, dim3(8192), dim3(256), 0, stream,
                       hsnew, co, hs, mask, out);
}
